// Attention_8735963480683
// MI455X (gfx1250) — matrix operand order repaired and checked by reading, not compiled, not run on hardware
//
#include <hip/hip_runtime.h>
#include <hip/hip_bf16.h>

typedef __attribute__((ext_vector_type(16))) __bf16 v16bf;
typedef __attribute__((ext_vector_type(8)))  __bf16 v8bf;
typedef __attribute__((ext_vector_type(8)))  float  v8f;
typedef __attribute__((ext_vector_type(4)))  float  v4f;

#define NB 32
#define NS 1024
#define ND 512
#define PROJ_ELEMS ((size_t)NB * NS * ND)          // 16,777,216 bf16 elements
#define PROJ_BYTES (PROJ_ELEMS * 2)                // 32 MiB per projection
#define LN_EPS 1e-5f

// ---------------------------------------------------------------------------
// Kernel A: convert + transpose the three weight matrices to bf16.
// wt layout: wt[mat][n][d] = W[mat][d][n]  (so B-fragment loads are contiguous)
// ---------------------------------------------------------------------------
__global__ __launch_bounds__(256) void prep_weights_kernel(
    const float* __restrict__ Wq, const float* __restrict__ Wk,
    const float* __restrict__ Wv, __bf16* __restrict__ wt) {
  int i = blockIdx.x * 256 + threadIdx.x;          // over 3*512*512
  int m = i >> 18;
  int n = (i >> 9) & 511;
  int d = i & 511;
  const float* W = (m == 0) ? Wq : ((m == 1) ? Wk : Wv);
  wt[i] = (__bf16)W[d * ND + n];
}

// ---------------------------------------------------------------------------
// Kernel B: fused QKV projection, relu(X @ W + b) -> bf16
// One wave computes a 16x64 output strip; K = 512 in 16 steps of 32.
// blockIdx = (rowBlock/4, nBlock, z) with z = seqIdx*3 + mat (6 combos).
// ---------------------------------------------------------------------------
__global__ __launch_bounds__(128) void proj_kernel(
    const float* __restrict__ seq1, const float* __restrict__ seq2,
    const __bf16* __restrict__ wt,
    const float* __restrict__ bq, const float* __restrict__ bk,
    const float* __restrict__ bv, __bf16* __restrict__ proj) {
  const int lane   = threadIdx.x & 31;
  const int waveId = threadIdx.x >> 5;
  const int z      = blockIdx.z;                   // 0..5
  const int seqIdx = z / 3, mat = z % 3;

  const float*  X    = seqIdx ? seq2 : seq1;
  const float*  bias = (mat == 0) ? bq : ((mat == 1) ? bk : bv);
  const __bf16* W    = wt + (size_t)mat * ND * ND;
  __bf16*       out  = proj + (size_t)z * PROJ_ELEMS;

  const int rowBase = (blockIdx.x * 4 + waveId) * 16;   // 0..32752
  const int n0      = blockIdx.y * 64;
  const int lr      = lane & 15;
  const bool hi     = lane >= 16;

  v8f acc[4] = {};
  const float* arow = X + (size_t)(rowBase + lr) * ND;

  for (int ks = 0; ks < 16; ++ks) {
    // A fragment (16x32, f32 -> bf16 on the fly), contiguous k order (same as B):
    // lanes 0-15: K = {ks*32+0..15}; lanes 16-31: K = {+16..31}
    const int kb = ks * 32 + (hi ? 16 : 0);
    v4f f0 = *(const v4f*)(arow + kb);
    v4f f1 = *(const v4f*)(arow + kb + 4);
    v4f f2 = *(const v4f*)(arow + kb + 8);
    v4f f3 = *(const v4f*)(arow + kb + 12);
    v16bf a;
#pragma unroll
    for (int i = 0; i < 4; ++i) {
      a[i]      = (__bf16)f0[i];
      a[4 + i]  = (__bf16)f1[i];
      a[8 + i]  = (__bf16)f2[i];
      a[12 + i] = (__bf16)f3[i];
    }
    // B fragments: lane n<16 -> col n, K ks*32+0..15 ; lanes 16-31 -> K +16..31
    const int kwb = ks * 32 + (hi ? 16 : 0);
#pragma unroll
    for (int nt = 0; nt < 4; ++nt) {
      const int col = n0 + nt * 16 + lr;
      v16bf bfr = *(const v16bf*)(W + (size_t)col * ND + kwb);
      acc[nt] = __builtin_amdgcn_wmma_f32_16x16x32_bf16(
          false, a, false, bfr, (short)0, acc[nt], false, false);
    }
  }
  // epilogue: bias + relu, store bf16.  C layout: lane<16 VGPR r = (M=r,N=lane),
  // lane>=16 VGPR r = (M=8+r, N=lane-16)
#pragma unroll
  for (int nt = 0; nt < 4; ++nt) {
    const int col  = n0 + nt * 16 + lr;
    const float bv_ = bias[col];
#pragma unroll
    for (int r = 0; r < 8; ++r) {
      const int row = rowBase + r + (hi ? 8 : 0);
      float v = fmaxf(acc[nt][r] + bv_, 0.0f);
      out[(size_t)row * ND + col] = (__bf16)v;
    }
  }
}

// ---------------------------------------------------------------------------
// Kernel C: cross scores.  a[b][s] = sum_t tanh( rowsrc[b,s,:] . colsrc[b,t,:] )
// dir 0: rowsrc=k1, colsrc=q2 -> a1 ; dir 1: rowsrc=k2, colsrc=q1 -> a2
// 4 waves/WG, each owns a 16-row block with A (16x512) resident in VGPRs.
// The 16x512 B tile (16 KB, contiguous) is staged into LDS once per WG via
// gfx1250 async copy (global_load_async_to_lds_b128, ASYNCcnt), double
// buffered so the next tile's DMA overlaps this tile's 16 WMMAs + tanh.
// ---------------------------------------------------------------------------
__device__ __forceinline__ float fast_tanh(float x) {
  float xc = fminf(fmaxf(x, -15.0f), 15.0f);       // tanh(+-15) == +-1 in fp32
  float e  = __expf(2.0f * xc);
  return (e - 1.0f) / (e + 1.0f);
}

// Copy one contiguous 16KB tile (16 rows x 512 bf16) to LDS asynchronously.
// 128 threads x 8 chunks x 16B, consecutive lanes -> consecutive 16B chunks.
__device__ __forceinline__ void stage_tile_async(const __bf16* __restrict__ gsrc,
                                                 unsigned lds_dst, int tid) {
#pragma unroll
  for (int i = 0; i < 8; ++i) {
    const int c = i * 128 + tid;                   // chunk 0..1023
    const __bf16* g = gsrc + (size_t)c * 8;        // 8 bf16 = 16B per chunk
    unsigned l = lds_dst + (unsigned)c * 16u;
    asm volatile("global_load_async_to_lds_b128 %0, %1, off"
                 :: "v"(l), "v"(g) : "memory");
  }
}

__global__ __launch_bounds__(128) void score_kernel(
    const __bf16* __restrict__ proj, float* __restrict__ araw) {
  const int tid    = threadIdx.x;
  const int lane   = tid & 31;
  const int waveId = tid >> 5;
  const int b      = blockIdx.y;
  const int dir    = blockIdx.z;

  const __bf16* rowsrc = proj + (size_t)(dir == 0 ? 1 : 4) * PROJ_ELEMS; // k1 / k2
  const __bf16* colsrc = proj + (size_t)(dir == 0 ? 3 : 0) * PROJ_ELEMS; // q2 / q1
  float* aout = araw + ((size_t)dir * NB + b) * NS;

  const int rowBlock = blockIdx.x * 4 + waveId;    // 0..63
  const int lr  = lane & 15;
  const bool hi = lane >= 16;

  __shared__ __align__(64) __bf16 btile[2][16 * ND];   // 2 x 16 KB

  // Load the 16 A k-fragments for this row block (stays resident).
  const __bf16* arow = rowsrc + ((size_t)b * NS + rowBlock * 16 + lr) * ND;
  v16bf A[16];
#pragma unroll
  for (int ks = 0; ks < 16; ++ks) {
    const int kb = ks * 32 + (hi ? 16 : 0);
    v8bf lo = *(const v8bf*)(arow + kb);
    v8bf up = *(const v8bf*)(arow + kb + 8);
#pragma unroll
    for (int i = 0; i < 8; ++i) { A[ks][i] = lo[i]; A[ks][8 + i] = up[i]; }
  }

  const __bf16* cbase = colsrc + (size_t)b * NS * ND;
  // LDS byte offset of btile: low 32 bits of the generic pointer (AS3 cast
  // places the LDS offset in the low dword).
  const unsigned lds0 = (unsigned)(uintptr_t)(&btile[0][0]);

  // Prologue: stage tile 0, wait, barrier.
  stage_tile_async(cbase, lds0, tid);
  asm volatile("s_wait_asynccnt 0x0" ::: "memory");
  __syncthreads();

  float rs[8] = {};
  for (int tt = 0; tt < 64; ++tt) {
    const int buf = tt & 1;
    if (tt + 1 < 64)                               // prefetch next tile
      stage_tile_async(cbase + (size_t)(tt + 1) * 16 * ND,
                       lds0 + (unsigned)(buf ^ 1) * (16u * ND * 2u), tid);

    // Compute this tile's 16x16 score block from LDS.
    const __bf16* brow = &btile[buf][(size_t)lr * ND + (hi ? 16 : 0)];
    v8f c = {};
#pragma unroll
    for (int ks = 0; ks < 16; ++ks) {
      v16bf bfr = *(const v16bf*)(brow + ks * 32);
      c = __builtin_amdgcn_wmma_f32_16x16x32_bf16(
          false, A[ks], false, bfr, (short)0, c, false, false);
    }
#pragma unroll
    for (int r = 0; r < 8; ++r) rs[r] += fast_tanh(c[r]);

    if (tt + 1 < 64)
      asm volatile("s_wait_asynccnt 0x0" ::: "memory");
    __syncthreads();   // prefetched tile visible + everyone done with 'buf'
  }

  // Row-sum across the 16 lanes of each half (rows 0..7 low, 8..15 high).
#pragma unroll
  for (int r = 0; r < 8; ++r) {
    float v = rs[r];
    v += __shfl_xor(v, 1);
    v += __shfl_xor(v, 2);
    v += __shfl_xor(v, 4);
    v += __shfl_xor(v, 8);
    rs[r] = v;
  }
  if (lr == 0) {
    const int base = rowBlock * 16 + (hi ? 8 : 0);
#pragma unroll
    for (int r = 0; r < 8; ++r) aout[base + r] = rs[r];
  }
}

// ---------------------------------------------------------------------------
// Kernel D: mask + softmax over S, weighted sum of V, mean(seq), layernorm.
// One 256-thread block per (batch, dir).
// ---------------------------------------------------------------------------
__global__ __launch_bounds__(256) void finalize_kernel(
    const float* __restrict__ araw,
    const int* __restrict__ mask1, const int* __restrict__ mask2,
    const __bf16* __restrict__ proj,
    const float* __restrict__ seq1, const float* __restrict__ seq2,
    const float* __restrict__ gamma, const float* __restrict__ beta,
    float* __restrict__ out) {
  const int b = blockIdx.x, dir = blockIdx.y, tid = threadIdx.x;
  const float*  a    = araw + ((size_t)dir * NB + b) * NS;
  const int*    mask = (dir ? mask2 : mask1) + (size_t)b * NS;
  const __bf16* v    = proj + (size_t)(dir ? 5 : 2) * PROJ_ELEMS + (size_t)b * NS * ND;
  const float*  seq  = (dir ? seq2 : seq1) + (size_t)b * NS * ND;

  __shared__ float p[NS];
  __shared__ float red[256];

  // masked scores + max
  float mx = -INFINITY;
#pragma unroll
  for (int i = 0; i < 4; ++i) {
    const int s = tid + i * 256;
    float val = mask[s] ? -INFINITY : a[s];
    p[s] = val;
    mx = fmaxf(mx, val);
  }
  red[tid] = mx; __syncthreads();
  for (int off = 128; off > 0; off >>= 1) {
    if (tid < off) red[tid] = fmaxf(red[tid], red[tid + off]);
    __syncthreads();
  }
  mx = red[0]; __syncthreads();

  // exp + sum
  float sum = 0.0f;
#pragma unroll
  for (int i = 0; i < 4; ++i) {
    const int s = tid + i * 256;
    float e = __expf(p[s] - mx);
    p[s] = e;
    sum += e;
  }
  red[tid] = sum; __syncthreads();
  for (int off = 128; off > 0; off >>= 1) {
    if (tid < off) red[tid] += red[tid + off];
    __syncthreads();
  }
  const float inv = 1.0f / red[0];
  __syncthreads();

  // weighted V sum + mean(seq) (each thread owns columns d0, d0+256)
  const int d0 = tid, d1 = tid + 256;
  float acc0 = 0.f, acc1 = 0.f, m0 = 0.f, m1 = 0.f;
  for (int s = 0; s < NS; ++s) {
    const float ps = p[s] * inv;
    const __bf16* vr = v + (size_t)s * ND;
    const float*  sr = seq + (size_t)s * ND;
    acc0 += ps * (float)vr[d0];
    acc1 += ps * (float)vr[d1];
    m0 += sr[d0];
    m1 += sr[d1];
  }
  const float x0 = m0 * (1.0f / (float)NS) + acc0;
  const float x1 = m1 * (1.0f / (float)NS) + acc1;

  // layernorm over 512
  red[tid] = x0 + x1; __syncthreads();
  for (int off = 128; off > 0; off >>= 1) {
    if (tid < off) red[tid] += red[tid + off];
    __syncthreads();
  }
  const float mu = red[0] * (1.0f / (float)ND); __syncthreads();
  red[tid] = (x0 - mu) * (x0 - mu) + (x1 - mu) * (x1 - mu); __syncthreads();
  for (int off = 128; off > 0; off >>= 1) {
    if (tid < off) red[tid] += red[tid + off];
    __syncthreads();
  }
  const float var  = red[0] * (1.0f / (float)ND);
  const float rinv = rsqrtf(var + LN_EPS);

  float* o = out + ((size_t)dir * NB + b) * ND;
  o[d0] = (x0 - mu) * rinv * gamma[d0] + beta[d0];
  o[d1] = (x1 - mu) * rinv * gamma[d1] + beta[d1];
}

// ---------------------------------------------------------------------------
extern "C" void kernel_launch(void* const* d_in, const int* in_sizes, int n_in,
                              void* d_out, int out_size, void* d_ws, size_t ws_size,
                              hipStream_t stream) {
  const float* seq1  = (const float*)d_in[0];
  const float* seq2  = (const float*)d_in[1];
  const int*   mask1 = (const int*)d_in[2];
  const int*   mask2 = (const int*)d_in[3];
  const float* Wq    = (const float*)d_in[4];
  const float* bq    = (const float*)d_in[5];
  const float* Wk    = (const float*)d_in[6];
  const float* bk    = (const float*)d_in[7];
  const float* Wv    = (const float*)d_in[8];
  const float* bv    = (const float*)d_in[9];
  const float* gamma = (const float*)d_in[10];
  const float* beta  = (const float*)d_in[11];

  char* ws = (char*)d_ws;
  __bf16* wt   = (__bf16*)ws;                           // 1.5 MiB
  __bf16* proj = (__bf16*)(ws + (size_t)(2 << 20));     // 6 * 32 MiB = 192 MiB
  float*  araw = (float*)(ws + (size_t)(2 << 20) + 6 * PROJ_BYTES); // 256 KiB

  prep_weights_kernel<<<dim3((3 * 512 * 512) / 256), 256, 0, stream>>>(Wq, Wk, Wv, wt);
  proj_kernel<<<dim3(512, 8, 6), 128, 0, stream>>>(seq1, seq2, wt, bq, bk, bv, proj);
  score_kernel<<<dim3(16, 32, 2), 128, 0, stream>>>(proj, araw);
  finalize_kernel<<<dim3(32, 2), 256, 0, stream>>>(araw, mask1, mask2, proj,
                                                   seq1, seq2, gamma, beta,
                                                   (float*)d_out);
}
